// RNN_Model_87866440941838
// MI455X (gfx1250) — compile-verified, matrix operand order checked by reading, not run on hardware
//
#include <hip/hip_runtime.h>
#include <math.h>

// ---------------------------------------------------------------------------
// RNN scan on gfx1250 (MI455X), bf16 WMMA path.
//   xp fused into the scan (saves ~0.5GB HBM traffic), persistent kernel with
//   one device-wide barrier per timestep, per-block weight slices in LDS.
//   v2: native v_tanh_f32, dual WMMA accumulator chains, emb prefetch.
// ---------------------------------------------------------------------------

typedef __attribute__((ext_vector_type(16))) __bf16 v16bf;
typedef __attribute__((ext_vector_type(8)))  __bf16 v8bf;
typedef __attribute__((ext_vector_type(8)))  float  v8f;

#define BB 128
#define SS 512
#define HH 1024
#define VV 128

// workspace layout (bytes, all 4KB aligned)
#define OFF_EMB  ((size_t)0)            // V*H bf16      = 262144
#define OFF_WIH  ((size_t)262144)       // H*H bf16      = 2097152
#define OFF_WHH  ((size_t)2359296)      // H*H bf16      = 2097152
#define OFF_WFC  ((size_t)4456448)      // V*H bf16      = 262144
#define OFF_BIAS ((size_t)4718592)      // H f32         = 4096
#define OFF_H0   ((size_t)4722688)      // B*H bf16      = 262144
#define OFF_H1   ((size_t)4984832)      // B*H bf16      = 262144
#define OFF_CNT  ((size_t)5246976)      // barrier counter

// Load 16 bf16 for one lane of a 16x32 A (or 32x16 B) tile:
// k..k+7 and k+16..k+23 (the lane's k-half offset is folded in by the caller).
__device__ __forceinline__ v16bf ld_tile16(const __bf16* p) {
    v8bf lo = *reinterpret_cast<const v8bf*>(p);
    v8bf hi = *reinterpret_cast<const v8bf*>(p + 16);
    return __builtin_shufflevector(lo, hi, 0,1,2,3,4,5,6,7,8,9,10,11,12,13,14,15);
}

__device__ __forceinline__ v8f wmma_bf16(v16bf a, v16bf b, v8f c) {
    return __builtin_amdgcn_wmma_f32_16x16x32_bf16(
        /*neg_a=*/false, a, /*neg_b=*/false, b,
        /*c_mod=*/(short)0, c, /*reuse_a=*/false, /*reuse_b=*/false);
}

// Branch-free tanh. gfx1250 has native V_TANH_F32; fallback is
// v_exp + v_rcp: tanh(x) = 1 - 2/(exp(2x)+1)  (exact at both saturations).
__device__ __forceinline__ float fast_tanh(float x) {
#if __has_builtin(__builtin_amdgcn_tanhf)
    return __builtin_amdgcn_tanhf(x);
#else
    float e = __expf(2.0f * x);
    return 1.0f - 2.0f * __builtin_amdgcn_rcpf(e + 1.0f);
#endif
}

// --------------------------- prep kernels ----------------------------------

__global__ void cvt_bf16_kernel(const float* __restrict__ in,
                                __bf16* __restrict__ out, int n) {
    int i = blockIdx.x * blockDim.x + threadIdx.x;
    if (i < n) out[i] = (__bf16)in[i];
}

__global__ void bias_sum_kernel(const float* __restrict__ bih,
                                const float* __restrict__ bhh,
                                float* __restrict__ out) {
    int i = blockIdx.x * blockDim.x + threadIdx.x;
    if (i < HH) out[i] = bih[i] + bhh[i];
}

// --------------------------- persistent scan -------------------------------
// grid = 64 blocks (one per 16-wide output-column tile), 256 threads = 8 waves
// (one wave per 16-row batch tile -> block covers the full [128 x 16] slice).

__global__ __launch_bounds__(256, 1)
void rnn_scan_kernel(const int*    __restrict__ x,
                     const __bf16* __restrict__ emb,
                     const __bf16* __restrict__ Wih,
                     const __bf16* __restrict__ Whh,
                     const float*  __restrict__ biassum,
                     __bf16* h0, __bf16* h1,
                     unsigned* counter) {
    const int n0      = blockIdx.x * 16;       // output column base
    const int wave    = threadIdx.x >> 5;
    const int lane    = threadIdx.x & 31;
    const int m0      = wave * 16;             // batch row base
    const int r15     = lane & 15;
    const int halfsel = lane >> 4;
    const int khalf   = halfsel * 8;

    // Stage this block's time-invariant weight slices into LDS (2 x 32KB).
    __shared__ __align__(16) unsigned short lds_ih[16 * HH];
    __shared__ __align__(16) unsigned short lds_hh[16 * HH];
    {
        const uint4* gih = reinterpret_cast<const uint4*>(Wih + (size_t)n0 * HH);
        const uint4* ghh = reinterpret_cast<const uint4*>(Whh + (size_t)n0 * HH);
        uint4* sih = reinterpret_cast<uint4*>(lds_ih);
        uint4* shh = reinterpret_cast<uint4*>(lds_hh);
        for (int c = threadIdx.x; c < (16 * HH) / 8; c += 256) {
            sih[c] = gih[c];
            shh[c] = ghh[c];
        }
    }
    __syncthreads();

    const float bias_v = biassum[n0 + r15];
    const __bf16* brow_ih = reinterpret_cast<const __bf16*>(lds_ih) + (size_t)r15 * HH;
    const __bf16* brow_hh = reinterpret_cast<const __bf16*>(lds_hh) + (size_t)r15 * HH;
    const int arow = m0 + r15;                 // A-matrix row this lane feeds

    for (int t = 0; t < SS; ++t) {
        const __bf16* hin  = (t & 1) ? h1 : h0;
        __bf16*       hout = (t & 1) ? h0 : h1;

        const int tok = x[(size_t)arow * SS + t];
        const __bf16* arow_e = emb + (size_t)tok * HH;
        const __bf16* arow_h = hin + (size_t)arow * HH;

        // Warm the WGP cache for next step's embedding gather (the only
        // data-dependent load in the loop). Emits global_prefetch_b8.
        if (t + 1 < SS) {
            const int tok_n = x[(size_t)arow * SS + (t + 1)];
            const __bf16* pn = emb + (size_t)tok_n * HH;
            __builtin_prefetch(pn + khalf, 0, 3);
            __builtin_prefetch(pn + khalf + 512, 0, 3);
        }

        // Two independent accumulator chains -> the xp-WMMA and h-WMMA
        // streams interleave instead of serializing through one C operand.
        v8f acc_x = {bias_v, bias_v, bias_v, bias_v, bias_v, bias_v, bias_v, bias_v};
        v8f acc_h = {0.f, 0.f, 0.f, 0.f, 0.f, 0.f, 0.f, 0.f};

        #pragma unroll 4
        for (int kk = 0; kk < HH; kk += 32) {
            const int ka = kk + khalf;
            v16bf ae = ld_tile16(arow_e + ka);     // emb[x[:,t]] tile  (global/L2)
            v16bf ah = ld_tile16(arow_h + ka);     // h tile            (global/L2)
            v16bf bi = ld_tile16(brow_ih + ka);    // W_ih slice        (LDS)
            v16bf bh = ld_tile16(brow_hh + ka);    // W_hh slice        (LDS)
            acc_x = wmma_bf16(ae, bi, acc_x);      // xp contribution
            acc_h = wmma_bf16(ah, bh, acc_h);      // recurrent contribution
        }

        // h_new = tanh(xp + h@Whh^T + bias); write bf16 for next step.
        #pragma unroll
        for (int r = 0; r < 8; ++r) {
            const int mrow = m0 + r + 8 * halfsel;
            hout[(size_t)mrow * HH + (n0 + r15)] =
                (__bf16)fast_tanh(acc_x[r] + acc_h[r]);
        }

        // device-wide barrier (monotonic counter; zeroed per launch)
        __syncthreads();
        if (threadIdx.x == 0) {
            __threadfence();
            __hip_atomic_fetch_add(counter, 1u, __ATOMIC_RELEASE,
                                   __HIP_MEMORY_SCOPE_AGENT);
            const unsigned target = (unsigned)(t + 1) * gridDim.x;
            while (__hip_atomic_load(counter, __ATOMIC_ACQUIRE,
                                     __HIP_MEMORY_SCOPE_AGENT) < target)
                __builtin_amdgcn_s_sleep(1);
        }
        __syncthreads();
    }
}

// --------------------------- final FC --------------------------------------
// out[b,v] = h_T[b,:] @ W_fc[v,:] + b_fc[v]; 8x8 tiles of 16x16 -> 64 waves.

__global__ __launch_bounds__(256, 1)
void fc_kernel(const __bf16* __restrict__ h,
               const __bf16* __restrict__ Wfc,
               const float*  __restrict__ bfc,
               float* __restrict__ out) {
    const int wave    = threadIdx.x >> 5;
    const int lane    = threadIdx.x & 31;
    const int tile    = blockIdx.x * 8 + wave;
    const int mt      = tile >> 3;
    const int nt      = tile & 7;
    const int r15     = lane & 15;
    const int halfsel = lane >> 4;
    const int khalf   = halfsel * 8;

    const __bf16* arow = h   + (size_t)(mt * 16 + r15) * HH;
    const __bf16* brow = Wfc + (size_t)(nt * 16 + r15) * HH;
    const float bv = bfc[nt * 16 + r15];
    v8f acc = {bv, bv, bv, bv, bv, bv, bv, bv};

    #pragma unroll 4
    for (int kk = 0; kk < HH; kk += 32) {
        v16bf a = ld_tile16(arow + kk + khalf);
        v16bf b = ld_tile16(brow + kk + khalf);
        acc = wmma_bf16(a, b, acc);
    }

    #pragma unroll
    for (int r = 0; r < 8; ++r) {
        const int mrow = mt * 16 + r + 8 * halfsel;
        out[(size_t)mrow * VV + nt * 16 + r15] = acc[r];
    }
}

// --------------------------- launch ----------------------------------------

extern "C" void kernel_launch(void* const* d_in, const int* in_sizes, int n_in,
                              void* d_out, int out_size, void* d_ws, size_t ws_size,
                              hipStream_t stream) {
    const int*   x    = (const int*)  d_in[0];
    const float* emb  = (const float*)d_in[1];
    const float* Wih  = (const float*)d_in[2];
    const float* Whh  = (const float*)d_in[3];
    const float* bih  = (const float*)d_in[4];
    const float* bhh  = (const float*)d_in[5];
    const float* Wfc  = (const float*)d_in[6];
    const float* bfc  = (const float*)d_in[7];

    char* ws = (char*)d_ws;
    __bf16*   emb_b = (__bf16*)(ws + OFF_EMB);
    __bf16*   wih_b = (__bf16*)(ws + OFF_WIH);
    __bf16*   whh_b = (__bf16*)(ws + OFF_WHH);
    __bf16*   wfc_b = (__bf16*)(ws + OFF_WFC);
    float*    biass = (float*) (ws + OFF_BIAS);
    __bf16*   h0    = (__bf16*)(ws + OFF_H0);
    __bf16*   h1    = (__bf16*)(ws + OFF_H1);
    unsigned* cnt   = (unsigned*)(ws + OFF_CNT);

    // deterministic init (graph-capture safe)
    hipMemsetAsync(h0,  0, (size_t)BB * HH * 2, stream);
    hipMemsetAsync(cnt, 0, 64, stream);

    // one-time bf16 conversions (L2-resident afterwards)
    cvt_bf16_kernel<<<(VV * HH + 255) / 256, 256, 0, stream>>>(emb, emb_b, VV * HH);
    cvt_bf16_kernel<<<(HH * HH + 255) / 256, 256, 0, stream>>>(Wih, wih_b, HH * HH);
    cvt_bf16_kernel<<<(HH * HH + 255) / 256, 256, 0, stream>>>(Whh, whh_b, HH * HH);
    cvt_bf16_kernel<<<(VV * HH + 255) / 256, 256, 0, stream>>>(Wfc, wfc_b, VV * HH);
    bias_sum_kernel<<<(HH + 255) / 256, 256, 0, stream>>>(bih, bhh, biass);

    // persistent scan: 64 blocks x 8 waves = full [128 x 1024] tile grid/step
    rnn_scan_kernel<<<HH / 16, 256, 0, stream>>>(x, emb_b, wih_b, whh_b,
                                                 biass, h0, h1, cnt);

    // final h lands in h0 (S even); FC -> d_out [128 x 128] f32
    fc_kernel<<<(BB / 16) * (VV / 16) / 8, 256, 0, stream>>>(
        h0, wfc_b, bfc, (float*)d_out);
}
